// Stage_GNN_learner_38190849196426
// MI455X (gfx1250) — matrix operand order repaired and checked by reading, not compiled, not run on hardware
//
#include <hip/hip_runtime.h>
#include <hip/hip_bf16.h>

// ---------------------------------------------------------------------------
// GNN graph-learner pipeline for MI455X (gfx1250, wave32, WMMA).
//   h    = relu(adj @ (X @ W1) + b1)
//   emb  = adj @ (h @ W2) + b2
//   e    = l2-normalize rows(emb);  S = e @ e^T
//   out  = relu(topk31_mask(S) * S)   (plus emb appended)
// Big GEMMs: bf16 WMMA (v_wmma_f32_16x16x32_bf16), f32 accumulate,
// software-pipelined (double-buffered) fragment loads.
// ---------------------------------------------------------------------------

typedef __bf16 bf16;
typedef __attribute__((ext_vector_type(16))) __bf16 bf16x16;
typedef __attribute__((ext_vector_type(8)))  __bf16 bf16x8;
typedef __attribute__((ext_vector_type(4)))  __bf16 bf16x4;
typedef __attribute__((ext_vector_type(8)))  float  f32x8;
typedef __attribute__((ext_vector_type(4)))  float  f32x4;

#define NN   8192   // nodes
#define FD   256    // feature dim
#define KEEP 31     // K+1

// ---------------- f32 -> bf16 conversion (vectorized) ----------------------
__global__ void cvt_f32_bf16_v4(const f32x4* __restrict__ src,
                                bf16x4* __restrict__ dst, long n4) {
  long i = (long)blockIdx.x * blockDim.x + threadIdx.x;
  if (i >= n4) return;
  f32x4 v = src[i];
  bf16x4 o;
  o.x = (bf16)v.x; o.y = (bf16)v.y; o.z = (bf16)v.z; o.w = (bf16)v.w;
  dst[i] = o;
}

// ---------------- transpose + convert (for W1/W2, 256x256) -----------------
__global__ void transpose_cvt(const float* __restrict__ src,
                              bf16* __restrict__ dst, int R, int C) {
  int c = blockIdx.x * 16 + threadIdx.x;
  int r = blockIdx.y * 16 + threadIdx.y;
  if (r < R && c < C) dst[(size_t)c * R + r] = (bf16)src[(size_t)r * C + c];
}

// ---------------- tiled WMMA GEMM ------------------------------------------
// C[M,N] = A[M,K] * B[K,N]  with B supplied transposed: BT[N,K] row-major.
// Both A and BT have leading dimension K.  K/32 must be even (true here).
// OUTMODE: 0 = f32, row-major C[M,N]
//          1 = bf16, row-major C[M,N]
//          2 = bf16, transposed C^T[N,M]  (feeds the next GEMM's BT operand)
// STREAM:  non-temporal f32 stores (for the 256MB S output, > L2 capacity).
// Block = 256 threads = 8 waves; block tile 128x128; wave tile 64x32.
template<int OUTMODE, bool BIAS, bool RELU, bool STREAM>
__global__ __launch_bounds__(256)
void gemm_bf16_wmma(const bf16* __restrict__ A, const bf16* __restrict__ BT,
                    float* __restrict__ Cf, bf16* __restrict__ Cb,
                    const float* __restrict__ bias, int M, int N, int K) {
  const int lane  = threadIdx.x & 31;
  const int wave  = threadIdx.x >> 5;
  const int wm    = wave & 1;     // 0..1  (M direction)
  const int wn    = wave >> 1;    // 0..3  (N direction)
  const int m0    = blockIdx.y * 128 + wm * 64;
  const int n0    = blockIdx.x * 128 + wn * 32;

  const int rsel  = lane & 15;    // row-within-16 selector
  const int khalf = lane >> 4;    // which K half this lane holds

  f32x8 acc[4][2];
#pragma unroll
  for (int i = 0; i < 4; ++i)
#pragma unroll
    for (int j = 0; j < 2; ++j) acc[i][j] = (f32x8){};

  // A fragment (16x32 bf16): lanes 0-15 hold K 0-7 & 16-23; lanes 16-31 hold
  // K 8-15 & 24-31 (ISA 7.12.2) -> two 16B loads at +khalf*8 and +khalf*8+16.
  const bf16* aptr[4];
#pragma unroll
  for (int i = 0; i < 4; ++i)
    aptr[i] = A + (size_t)(m0 + 16 * i + rsel) * K + khalf * 8;

  // B fragment (32x16 bf16): lane = N col; lanes 0-15 hold K 0-7 & 16-23,
  // lanes 16-31 hold K 8-15 & 24-31 -> two 16B loads at +khalf*8 and +khalf*8+16.
  const bf16* bptr[2];
#pragma unroll
  for (int j = 0; j < 2; ++j)
    bptr[j] = BT + (size_t)(n0 + 16 * j + rsel) * K + khalf * 8;

  auto loadA = [&](int i, int k) {
    union { bf16x16 v; bf16x8 h[2]; } u;
    u.h[0] = *(const bf16x8*)(aptr[i] + k);
    u.h[1] = *(const bf16x8*)(aptr[i] + k + 16);
    return u.v;
  };
  auto loadB = [&](int j, int k) {
    union { bf16x16 v; bf16x8 h[2]; } u;
    u.h[0] = *(const bf16x8*)(bptr[j] + k);
    u.h[1] = *(const bf16x8*)(bptr[j] + k + 16);
    return u.v;
  };

  bf16x16 A0[4], B0[2], A1[4], B1[2];
  auto mma = [&](bf16x16 (&a)[4], bf16x16 (&b)[2]) {
#pragma unroll
    for (int i = 0; i < 4; ++i)
#pragma unroll
      for (int j = 0; j < 2; ++j)
        acc[i][j] = __builtin_amdgcn_wmma_f32_16x16x32_bf16(
            false, a[i], false, b[j], (short)0, acc[i][j], false, false);
  };

  // Software pipeline, 2-deep, 2x unrolled.  S = K/32 (even, >= 2).
  const int S = K >> 5;
#pragma unroll
  for (int i = 0; i < 4; ++i) A0[i] = loadA(i, 0);
#pragma unroll
  for (int j = 0; j < 2; ++j) B0[j] = loadB(j, 0);

  int k = 32;
  for (int s = 0; s < S - 2; s += 2, k += 64) {
#pragma unroll
    for (int i = 0; i < 4; ++i) A1[i] = loadA(i, k);
#pragma unroll
    for (int j = 0; j < 2; ++j) B1[j] = loadB(j, k);
    mma(A0, B0);
#pragma unroll
    for (int i = 0; i < 4; ++i) A0[i] = loadA(i, k + 32);
#pragma unroll
    for (int j = 0; j < 2; ++j) B0[j] = loadB(j, k + 32);
    mma(A1, B1);
  }
#pragma unroll
  for (int i = 0; i < 4; ++i) A1[i] = loadA(i, k);
#pragma unroll
  for (int j = 0; j < 2; ++j) B1[j] = loadB(j, k);
  mma(A0, B0);
  mma(A1, B1);

  // Epilogue. C/D layout: VGPR v, lanes 0-15 -> (M=v, N=lane),
  // lanes 16-31 -> (M=v+8, N=lane-16).
  const int nlocal = lane & 15;
  const int moff   = (lane >> 4) * 8;
#pragma unroll
  for (int j = 0; j < 2; ++j) {
    const int col = n0 + 16 * j + nlocal;
    const float bv = BIAS ? bias[col] : 0.0f;
#pragma unroll
    for (int i = 0; i < 4; ++i) {
      const int rowbase = m0 + 16 * i + moff;
#pragma unroll
      for (int v = 0; v < 8; ++v) {
        float x = acc[i][j][v] + bv;
        if (RELU) x = fmaxf(x, 0.0f);
        const int row = rowbase + v;
        if (OUTMODE == 0) {
          float* p = &Cf[(size_t)row * N + col];
          if (STREAM) __builtin_nontemporal_store(x, p);
          else        *p = x;
        } else if (OUTMODE == 1) {
          Cb[(size_t)row * N + col] = (bf16)x;
        } else {
          Cb[(size_t)col * M + row] = (bf16)x;
        }
      }
    }
  }
}

// ---------------- row-wise L2 normalize: f32 emb -> bf16 e -----------------
__global__ __launch_bounds__(256)
void rownorm_bf16(const float* __restrict__ emb, bf16* __restrict__ e) {
  __shared__ float red[256];
  const int row = blockIdx.x;
  const float v = emb[(size_t)row * FD + threadIdx.x];
  red[threadIdx.x] = v * v;
  __syncthreads();
  for (int s = 128; s > 0; s >>= 1) {
    if ((int)threadIdx.x < s) red[threadIdx.x] += red[threadIdx.x + s];
    __syncthreads();
  }
  const float scale = 1.0f / fmaxf(sqrtf(red[0]), 1e-12f);
  e[(size_t)row * FD + threadIdx.x] = (bf16)(v * scale);
}

// ---------------- exact top-31 per row + relu, in place --------------------
__global__ __launch_bounds__(256)
void topk_relu(float* __restrict__ S, int Ncols, int keep) {
  const int T = 256;
  const int PER = Ncols / T;           // 32
  __shared__ float smax[256];
  __shared__ int   sidx[256];
  const int tid = threadIdx.x;
  float* rp = S + (size_t)blockIdx.x * Ncols;

  float vals[32], work[32];
#pragma unroll
  for (int p = 0; p < 32; ++p) {
    vals[p] = __builtin_nontemporal_load(&rp[tid + p * T]);
    work[p] = vals[p];
  }

  float thr = -3.0e38f;
  for (int it = 0; it < keep; ++it) {
    float lm = -3.0e38f; int li = 0;
#pragma unroll
    for (int p = 0; p < 32; ++p)
      if (work[p] > lm) { lm = work[p]; li = p; }
    smax[tid] = lm;
    sidx[tid] = tid * PER + li;
    __syncthreads();
    for (int s = 128; s > 0; s >>= 1) {
      if (tid < s) {
        const bool take = (smax[tid + s] > smax[tid]) ||
                          (smax[tid + s] == smax[tid] &&
                           sidx[tid + s] < sidx[tid]);
        if (take) { smax[tid] = smax[tid + s]; sidx[tid] = sidx[tid + s]; }
      }
      __syncthreads();
    }
    thr = smax[0];
    const int widx = sidx[0];
    __syncthreads();
    if (widx / PER == tid) work[widx % PER] = -3.0e38f;
  }

#pragma unroll
  for (int p = 0; p < 32; ++p) {
    const float v = vals[p];
    __builtin_nontemporal_store((v >= thr && v > 0.0f) ? v : 0.0f,
                                &rp[tid + p * T]);
  }
}

// ---------------------------------------------------------------------------
extern "C" void kernel_launch(void* const* d_in, const int* in_sizes, int n_in,
                              void* d_out, int out_size, void* d_ws,
                              size_t ws_size, hipStream_t stream) {
  const float* features = (const float*)d_in[0];  // [NN, FD]
  const float* adj      = (const float*)d_in[1];  // [NN, NN]
  const float* W1       = (const float*)d_in[2];  // [FD, FD]
  const float* b1       = (const float*)d_in[3];  // [FD]
  const float* W2       = (const float*)d_in[4];  // [FD, FD]
  const float* b2       = (const float*)d_in[5];  // [FD]

  float* Sout = (float*)d_out;                         // [NN, NN]
  float* Emb  = Sout + (size_t)NN * NN;                // [NN, FD]

  // Workspace carve-up (all bf16, ~154 MB total).
  char* w = (char*)d_ws;
  bf16* Ab  = (bf16*)w; w += (size_t)NN * NN * sizeof(bf16);   // adj
  bf16* Xb  = (bf16*)w; w += (size_t)NN * FD * sizeof(bf16);   // features
  bf16* W1T = (bf16*)w; w += (size_t)FD * FD * sizeof(bf16);
  bf16* W2T = (bf16*)w; w += (size_t)FD * FD * sizeof(bf16);
  bf16* S1T = (bf16*)w; w += (size_t)NN * FD * sizeof(bf16);   // (X@W1)^T
  bf16* Hb  = (bf16*)w; w += (size_t)NN * FD * sizeof(bf16);   // h
  bf16* S2T = (bf16*)w; w += (size_t)NN * FD * sizeof(bf16);   // (h@W2)^T
  bf16* Eb  = (bf16*)w; w += (size_t)NN * FD * sizeof(bf16);   // normalized emb

  // 1) Convert hot operands to bf16.
  {
    long n4 = (long)NN * FD / 4;
    cvt_f32_bf16_v4<<<(unsigned)((n4 + 255) / 256), 256, 0, stream>>>(
        (const f32x4*)features, (bf16x4*)Xb, n4);
    n4 = (long)NN * NN / 4;
    cvt_f32_bf16_v4<<<(unsigned)((n4 + 255) / 256), 256, 0, stream>>>(
        (const f32x4*)adj, (bf16x4*)Ab, n4);
    transpose_cvt<<<dim3(FD / 16, FD / 16), dim3(16, 16), 0, stream>>>(
        W1, W1T, FD, FD);
    transpose_cvt<<<dim3(FD / 16, FD / 16), dim3(16, 16), 0, stream>>>(
        W2, W2T, FD, FD);
  }

  const dim3 blk(256);
  const dim3 grid_skinny(FD / 128, NN / 128);   // (2, 64)
  const dim3 grid_square(NN / 128, NN / 128);   // (64, 64)

  // 2) S1^T = (X @ W1)^T                      [FD x NN] bf16
  gemm_bf16_wmma<2, false, false, false><<<grid_skinny, blk, 0, stream>>>(
      Xb, W1T, nullptr, S1T, nullptr, NN, FD, FD);

  // 3) H = relu(adj @ S1 + b1)                [NN x FD] bf16
  gemm_bf16_wmma<1, true, true, false><<<grid_skinny, blk, 0, stream>>>(
      Ab, S1T, nullptr, Hb, b1, NN, FD, NN);

  // 4) S2^T = (H @ W2)^T                      [FD x NN] bf16
  gemm_bf16_wmma<2, false, false, false><<<grid_skinny, blk, 0, stream>>>(
      Hb, W2T, nullptr, S2T, nullptr, NN, FD, FD);

  // 5) EMB = adj @ S2 + b2                    [NN x FD] f32 (output)
  gemm_bf16_wmma<0, true, false, false><<<grid_skinny, blk, 0, stream>>>(
      Ab, S2T, Emb, nullptr, b2, NN, FD, NN);

  // 6) e = l2-normalize rows(EMB)             [NN x FD] bf16
  rownorm_bf16<<<NN, 256, 0, stream>>>(Emb, Eb);

  // 7) S = e @ e^T  (BT operand is e itself)  [NN x NN] f32 (output)
  gemm_bf16_wmma<0, false, false, true><<<grid_square, blk, 0, stream>>>(
      Eb, Eb, Sout, nullptr, nullptr, NN, NN, FD);

  // 8) exact top-31 per row + relu, in place on S.
  topk_relu<<<NN, 256, 0, stream>>>(Sout, NN, KEEP);
}
